// EnhancedBlock_71193377899073
// MI455X (gfx1250) — matrix operand order repaired and checked by reading, not compiled, not run on hardware
//
#include <hip/hip_runtime.h>
#include <math.h>

#define Bsz 4
#define Tn  1024
#define Cn  768
#define Hn  12
#define HDn 64
#define C3  2304
#define C4  3072

typedef __attribute__((ext_vector_type(16))) __bf16 v16bf;
typedef __attribute__((ext_vector_type(2)))  __bf16 v2bf;
typedef __attribute__((ext_vector_type(8)))  float  v8f;

union bfu { unsigned u[8]; v16bf v; };

__device__ __forceinline__ unsigned short f2bf(float f) {
  __bf16 h = (__bf16)f;                       // native v_cvt f32->bf16
  return __builtin_bit_cast(unsigned short, h);
}
// build_vector(fptrunc,fptrunc) -> single v_cvt_pk_bf16_f32
__device__ __forceinline__ unsigned pack2(float a, float b) {
  v2bf t;
  t.x = (__bf16)a;
  t.y = (__bf16)b;
  return __builtin_bit_cast(unsigned, t);
}
__device__ __forceinline__ float sigf(float x) { return 1.0f / (1.0f + __expf(-x)); }
__device__ __forceinline__ float softplusf(float x) {
  return (x > 20.f) ? x : log1pf(__expf(x));
}

// ---------------------------------------------------------------------------
// Generic batched GEMM:  D = epilogue( A[M,K] x W[K,N] + bias (+R) )
// REQUIRES M%64==0, N%64==0, K%32==0 (all launches satisfy this).
// A,W fp32 in global, converted to bf16 into LDS, WMMA f32_16x16x32_bf16.
// wT: W stored [N][K].  Batch over blockIdx.z with strides.
// ep: 0 none, 1 relu, 2 sigmoid, 3 silu, 4 plastic, 5 plastic+gelu,
//     6 sigmoid(acc+bias)*R
// ---------------------------------------------------------------------------
__global__ __launch_bounds__(128) void k_gemm(
    const float* __restrict__ A, long aB, int lda,
    const float* __restrict__ W, long wB, int ldw, int wT,
    const float* __restrict__ bias, long biasB,
    const float* __restrict__ R, long rB, int ldr,
    float* __restrict__ D, long dB, int ldd,
    int M, int N, int K, int ep, const float* __restrict__ s0)
{
  __shared__ unsigned As[64 * 32 / 2];   // [m][k] bf16 pairs
  __shared__ unsigned Bs[64 * 32 / 2];   // [n][k] bf16 pairs (B transposed)
  unsigned short* Bsh = (unsigned short*)Bs;

  int z = blockIdx.z;
  A += (long)z * aB;
  W += (long)z * wB;
  D += (long)z * dB;
  if (R) R += (long)z * rB;
  if (bias) bias += (long)z * biasB;

  int m0 = blockIdx.y * 64, n0 = blockIdx.x * 64;
  int tid = threadIdx.x;
  int wv = tid >> 5, lane = tid & 31;
  int g = lane >> 4, lr = lane & 15;
  int wm = (wv >> 1) * 32, wn = (wv & 1) * 32;

  v8f zz = {0.f, 0.f, 0.f, 0.f, 0.f, 0.f, 0.f, 0.f};
  v8f acc[2][2];
  acc[0][0] = zz; acc[0][1] = zz; acc[1][0] = zz; acc[1][1] = zz;

  for (int k0 = 0; k0 < K; k0 += 32) {
    // ---- stage A tile (64x32) : 512 float4 loads, 4 per thread ----
#pragma unroll
    for (int e = 0; e < 4; ++e) {
      int vi = e * 128 + tid;
      int mm = vi >> 3;              // 8 float4 per row
      int c4 = (vi & 7) * 4;
      const float4 a4 =
          *(const float4*)(A + (long)(m0 + mm) * lda + k0 + c4);
      As[(mm * 32 + c4) >> 1]       = pack2(a4.x, a4.y);
      As[(mm * 32 + c4 + 2) >> 1]   = pack2(a4.z, a4.w);
    }
    // ---- stage W tile into Bs[n][k] ----
    if (wT) {                        // W[n][k]: same pattern as A
#pragma unroll
      for (int e = 0; e < 4; ++e) {
        int vi = e * 128 + tid;
        int nn = vi >> 3;
        int c4 = (vi & 7) * 4;
        const float4 b4 =
            *(const float4*)(W + (long)(n0 + nn) * ldw + k0 + c4);
        Bs[(nn * 32 + c4) >> 1]     = pack2(b4.x, b4.y);
        Bs[(nn * 32 + c4 + 2) >> 1] = pack2(b4.z, b4.w);
      }
    } else {                         // W[k][n]: vector load along n, scatter
#pragma unroll
      for (int e = 0; e < 4; ++e) {
        int vi = e * 128 + tid;
        int kk = vi >> 4;            // 16 float4 per k row
        int n4 = (vi & 15) * 4;
        const float4 b4 =
            *(const float4*)(W + (long)(k0 + kk) * ldw + n0 + n4);
        Bsh[(n4 + 0) * 32 + kk] = f2bf(b4.x);
        Bsh[(n4 + 1) * 32 + kk] = f2bf(b4.y);
        Bsh[(n4 + 2) * 32 + kk] = f2bf(b4.z);
        Bsh[(n4 + 3) * 32 + kk] = f2bf(b4.w);
      }
    }
    __syncthreads();

    bfu a[2], b[2];
#pragma unroll
    for (int i = 0; i < 2; ++i) {
      int mr = wm + i * 16 + lr;
#pragma unroll
      for (int v = 0; v < 8; ++v) {
        int kk = g * 16 + v * 2;
        a[i].u[v] = As[(mr * 32 + kk) >> 1];
      }
    }
#pragma unroll
    for (int j = 0; j < 2; ++j) {
      int nr = wn + j * 16 + lr;
      int kb = g * 16;
#pragma unroll
      for (int v = 0; v < 8; ++v)
        b[j].u[v] = Bs[(nr * 32 + kb + 2 * v) >> 1];
    }
#pragma unroll
    for (int i = 0; i < 2; ++i)
#pragma unroll
      for (int j = 0; j < 2; ++j)
        acc[i][j] = __builtin_amdgcn_wmma_f32_16x16x32_bf16(
            false, a[i].v, false, b[j].v, (short)0, acc[i][j], false, false);
    __syncthreads();
  }

  float sc = s0 ? s0[0] : 0.f;
#pragma unroll
  for (int i = 0; i < 2; ++i)
#pragma unroll
    for (int j = 0; j < 2; ++j)
#pragma unroll
      for (int r = 0; r < 8; ++r) {
        int mi = m0 + wm + i * 16 + r + g * 8;
        int ni = n0 + wn + j * 16 + lr;
        float v = acc[i][j][r];
        if (bias) v += bias[ni];
        if (ep == 6) {
          v = sigf(v) * (R ? R[(long)mi * ldr + ni] : 1.f);
        } else {
          if (R) v += R[(long)mi * ldr + ni];
          if (ep == 1) v = fmaxf(v, 0.f);
          else if (ep == 2) v = sigf(v);
          else if (ep == 3) v = v * sigf(v);
          else if (ep == 4 || ep == 5) {
            float a1 = softplusf(v);
            float a2 = a1 * a1;
            v = a2 / (a2 + 0.250001f) * (1.f + sc);
            if (ep == 5) v = 0.5f * v * (1.f + erff(v * 0.70710678f));
          }
        }
        D[(long)mi * ldd + ni] = v;
      }
}

// ---------------------------------------------------------------------------
// Flash attention (wave32 WMMA): one wave = 16 queries, online softmax over
// 32-key blocks.  qm/km/vv are (B*H, T, HD) contiguous. rb = rel_bias (H,T,T).
// ---------------------------------------------------------------------------
__global__ __launch_bounds__(128) void k_attn(
    const float* __restrict__ qm, const float* __restrict__ km,
    const float* __restrict__ vv, const float* __restrict__ rb,
    float* __restrict__ outp)
{
  __shared__ unsigned Pt[4][256];     // per-wave 16x32 bf16 P tile
  int wv = threadIdx.x >> 5, lane = threadIdx.x & 31;
  int g = lane >> 4, lr = lane & 15;
  unsigned short* myP = (unsigned short*)Pt[wv];
  unsigned* myP32 = Pt[wv];
  int z = blockIdx.y;
  int h = z % Hn;
  int q0 = blockIdx.x * 64 + wv * 16;

  v8f zz = {0.f, 0.f, 0.f, 0.f, 0.f, 0.f, 0.f, 0.f};

  const float* qb = qm + ((long)z * Tn + q0) * HDn;
  bfu qa[2];
#pragma unroll
  for (int s = 0; s < 2; ++s)
#pragma unroll
    for (int v = 0; v < 8; ++v) {
      int kk = g * 16 + v * 2;
      int hd = s * 32 + kk;
      qa[s].u[v] = pack2(qb[lr * HDn + hd], qb[lr * HDn + hd + 1]);
    }

  v8f acc[4];
  acc[0] = zz; acc[1] = zz; acc[2] = zz; acc[3] = zz;
  float rmax[8], rsum[8];
#pragma unroll
  for (int r = 0; r < 8; ++r) { rmax[r] = -1e30f; rsum[r] = 0.f; }

  const float* bb = rb + ((long)h * Tn + q0) * Tn;

  for (int kb = 0; kb < Tn; kb += 32) {
    float p[2][8];
#pragma unroll
    for (int kt = 0; kt < 2; ++kt) {
      const float* kp = km + ((long)z * Tn + kb + kt * 16) * HDn;
      bfu kf[2];
#pragma unroll
      for (int s = 0; s < 2; ++s)
#pragma unroll
        for (int v = 0; v < 8; ++v) {
          int hd = s * 32 + g * 16 + 2 * v;
          kf[s].u[v] = pack2(kp[lr * HDn + hd], kp[lr * HDn + hd + 1]);
        }
      v8f st = zz;
      st = __builtin_amdgcn_wmma_f32_16x16x32_bf16(false, qa[0].v, false, kf[0].v,
                                                   (short)0, st, false, false);
      st = __builtin_amdgcn_wmma_f32_16x16x32_bf16(false, qa[1].v, false, kf[1].v,
                                                   (short)0, st, false, false);
#pragma unroll
      for (int r = 0; r < 8; ++r)
        p[kt][r] = st[r] * 0.125f + bb[(long)(r + g * 8) * Tn + kb + kt * 16 + lr];
    }
#pragma unroll
    for (int r = 0; r < 8; ++r) {
      float t = fmaxf(p[0][r], p[1][r]);
      for (int m = 1; m < 16; m <<= 1) t = fmaxf(t, __shfl_xor(t, m, 32));
      float nm = fmaxf(rmax[r], t);
      float sc = __expf(rmax[r] - nm);
      rmax[r] = nm;
      float p0 = __expf(p[0][r] - nm), p1 = __expf(p[1][r] - nm);
      p[0][r] = p0; p[1][r] = p1;
      float ts = p0 + p1;
      for (int m = 1; m < 16; m <<= 1) ts += __shfl_xor(ts, m, 32);
      rsum[r] = rsum[r] * sc + ts;
#pragma unroll
      for (int nf = 0; nf < 4; ++nf) acc[nf][r] *= sc;
    }
    // write P tile (two bf16 halves per row entry)
#pragma unroll
    for (int r = 0; r < 8; ++r) {
      myP[(r + g * 8) * 32 + lr]      = f2bf(p[0][r]);
      myP[(r + g * 8) * 32 + 16 + lr] = f2bf(p[1][r]);
    }
    __syncthreads();
    bfu pa;
#pragma unroll
    for (int v = 0; v < 8; ++v) {
      int kk = g * 16 + v * 2;
      pa.u[v] = myP32[(lr * 32 + kk) >> 1];
    }
    const float* vp = vv + ((long)z * Tn + kb) * HDn;
#pragma unroll
    for (int nf = 0; nf < 4; ++nf) {
      bfu vf;
      int n = nf * 16 + lr;
#pragma unroll
      for (int v = 0; v < 8; ++v) {
        int kk = g * 16 + 2 * v;
        vf.u[v] = pack2(vp[(long)kk * HDn + n], vp[(long)(kk + 1) * HDn + n]);
      }
      acc[nf] = __builtin_amdgcn_wmma_f32_16x16x32_bf16(
          false, pa.v, false, vf.v, (short)0, acc[nf], false, false);
    }
    __syncthreads();
  }
  float* op = outp + ((long)z * Tn + q0) * HDn;
#pragma unroll
  for (int nf = 0; nf < 4; ++nf)
#pragma unroll
    for (int r = 0; r < 8; ++r)
      op[(long)(r + g * 8) * HDn + nf * 16 + lr] = acc[nf][r] / rsum[r];
}

// ---------------------------------------------------------------------------
// LayerNorm (+ optional feedback add). One 256-thread block per row of C=768.
// ---------------------------------------------------------------------------
__global__ __launch_bounds__(256) void k_ln(
    const float* __restrict__ x, const float* __restrict__ fbp,
    const float* __restrict__ g, const float* __restrict__ b,
    float* __restrict__ out)
{
  __shared__ float red[256];
  long row = blockIdx.x;
  int tid = threadIdx.x;
  const float* xr = x + row * Cn;
  float s = 0.f, s2 = 0.f;
#pragma unroll
  for (int i = 0; i < 3; ++i) {
    float v = xr[tid + i * 256];
    s += v; s2 += v * v;
  }
  red[tid] = s; __syncthreads();
  for (int o = 128; o > 0; o >>= 1) { if (tid < o) red[tid] += red[tid + o]; __syncthreads(); }
  float mean = red[0] / (float)Cn; __syncthreads();
  red[tid] = s2; __syncthreads();
  for (int o = 128; o > 0; o >>= 1) { if (tid < o) red[tid] += red[tid + o]; __syncthreads(); }
  float var = red[0] / (float)Cn - mean * mean;
  float rstd = rsqrtf(var + 1e-5f);
#pragma unroll
  for (int i = 0; i < 3; ++i) {
    int c = tid + i * 256;
    float y = (xr[c] - mean) * rstd * g[c] + b[c];
    if (fbp) y += fbp[row * Cn + c];
    out[row * Cn + c] = y;
  }
}

// qkv (B,T,3C) -> q,k,v each (B*H, T, HD) contiguous
__global__ void k_split(const float* __restrict__ qkv, float* __restrict__ q,
                        float* __restrict__ k, float* __restrict__ v)
{
  long i = (long)blockIdx.x * 256 + threadIdx.x;
  long tot = (long)Bsz * Tn * Cn;
  if (i >= tot) return;
  int b = (int)(i / ((long)Tn * Cn));
  int rem = (int)(i % ((long)Tn * Cn));
  int t = rem / Cn, cc = rem % Cn;
  int h = cc / HDn, d = cc % HDn;
  long dst = (((long)b * Hn + h) * Tn + t) * HDn + d;
  long src = ((long)b * Tn + t) * C3 + cc;
  q[dst] = qkv[src];
  k[dst] = qkv[src + Cn];
  v[dst] = qkv[src + 2 * Cn];
}

// (B,H,T,HD) -> (B,T,C)
__global__ void k_perm(const float* __restrict__ fh, float* __restrict__ fo)
{
  long i = (long)blockIdx.x * 256 + threadIdx.x;
  long tot = (long)Bsz * Tn * Cn;
  if (i >= tot) return;
  int b = (int)(i / ((long)Tn * Cn));
  int rem = (int)(i % ((long)Tn * Cn));
  int t = rem / Cn, cc = rem % Cn;
  int h = cc / HDn, d = cc % HDn;
  fo[i] = fh[(((long)b * Hn + h) * Tn + t) * HDn + d];
}

// circulant kernel values: ker[d] = irfft(exp(-f*alpha))[d]
__global__ __launch_bounds__(1024) void k_ker(const float* __restrict__ alpha,
                                              float* __restrict__ ker)
{
  int d = threadIdx.x;
  float a = alpha[0];
  float s = 1.0f;                              // f = 0 term
  for (int f = 1; f < 512; ++f) {
    int fd = (f * d) & 1023;
    s += 2.f * __expf(-(float)f * a) * __cosf(6.28318530718f * (float)fd / 1024.f);
  }
  s += __expf(-512.f * a) * ((d & 1) ? -1.f : 1.f);
  ker[d] = s / 1024.f;
}

__global__ void k_circ(const float* __restrict__ ker, float* __restrict__ K)
{
  int i = blockIdx.x * 256 + threadIdx.x;
  int t = i >> 10, ss = i & 1023;
  K[i] = ker[(t - ss) & 1023];
}

// latb[z,n] = (mean_t v[z,t,:]) @ W_lat + b_lat, then @ W_fuse[64:] + b_fuse
__global__ __launch_bounds__(64) void k_latent(
    const float* __restrict__ v, const float* __restrict__ Wlat,
    const float* __restrict__ blat, const float* __restrict__ Wfuse,
    const float* __restrict__ bfuse, float* __restrict__ latb)
{
  __shared__ float mn[HDn], lt[HDn];
  int z = blockIdx.x, d = threadIdx.x;
  float s = 0.f;
  for (int t = 0; t < Tn; ++t) s += v[((long)z * Tn + t) * HDn + d];
  mn[d] = s / (float)Tn;
  __syncthreads();
  float a = blat[d];
  for (int j = 0; j < HDn; ++j) a += mn[j] * Wlat[j * HDn + d];
  lt[d] = a;
  __syncthreads();
  float o = bfuse[d];
  for (int j = 0; j < HDn; ++j) o += lt[j] * Wfuse[(HDn + j) * HDn + d];
  latb[(long)z * HDn + d] = o;
}

// GRU recurrence, one block per batch element (serial over T)
__global__ __launch_bounds__(1024) void k_gru(
    const float* __restrict__ xg, const float* __restrict__ Whh,
    const float* __restrict__ bhh, float* __restrict__ out)
{
  __shared__ float hs[Cn];
  __shared__ float gh[C3];
  int b = blockIdx.x, tid = threadIdx.x;
  for (int c = tid; c < Cn; c += 1024) hs[c] = 0.f;
  __syncthreads();
  for (int t = 0; t < Tn; ++t) {
    const float* xgt = xg + ((long)b * Tn + t) * C3;
    for (int j = tid; j < C3; j += 1024) {
      float s = bhh[j];
      const float* wr = Whh + (long)j * Cn;
      for (int c = 0; c < Cn; ++c) s += hs[c] * wr[c];
      gh[j] = s;
    }
    __syncthreads();
    for (int c = tid; c < Cn; c += 1024) {
      float r  = sigf(xgt[c] + gh[c]);
      float zt = sigf(xgt[Cn + c] + gh[Cn + c]);
      float n  = tanhf(xgt[2 * Cn + c] + r * gh[2 * Cn + c]);
      float h2 = (1.f - zt) * n + zt * hs[c];
      out[((long)b * Tn + t) * Cn + c] = h2;
      hs[c] = h2;
    }
    __syncthreads();
  }
}

// x3 = x2 + mlp_out * (gating * exp(-te) * stdp * meta)
__global__ void k_stdp(
    const float* __restrict__ x2, const float* __restrict__ mlpin,
    const float* __restrict__ prev, const float* __restrict__ mlpout,
    const float* __restrict__ gat, const float* __restrict__ convw,
    const float* __restrict__ convb, const float* __restrict__ stdp,
    const float* __restrict__ meta, float* __restrict__ x3)
{
  long i = (long)blockIdx.x * 256 + threadIdx.x;
  long tot = (long)Bsz * Tn * Cn;
  if (i >= tot) return;
  int t = (int)((i / Cn) % Tn);
  float d0 = (t > 0)      ? fabsf(mlpin[i - Cn] - prev[i - Cn]) : 0.f;
  float d1 =                fabsf(mlpin[i]      - prev[i]);
  float d2 = (t < Tn - 1) ? fabsf(mlpin[i + Cn] - prev[i + Cn]) : 0.f;
  float te = convw[0] * d0 + convw[1] * d1 + convw[2] * d2 + convb[0];
  float gg = gat[i] * __expf(-te) * stdp[0] * meta[0];
  x3[i] = x2[i] + mlpout[i] * gg;
}

__global__ void k_osc(
    const float* __restrict__ x3, const float* __restrict__ amp,
    const float* __restrict__ freq, const float* __restrict__ phase,
    const float* __restrict__ mu, const float* __restrict__ sigma,
    float* __restrict__ modb)
{
  long i = (long)blockIdx.x * 256 + threadIdx.x;
  long tot = (long)Bsz * Tn * Cn;
  if (i >= tot) return;
  float xc = x3[i] - mu[0];
  float sg = sigma[0];
  modb[i] = amp[0] * __expf(-xc * xc / (2.f * sg * sg)) *
            __cosf(6.28318530718f * freq[0] * xc + phase[0]);
}

// top-k(384) threshold via bisection, normalize, standardize (ddof=1)
__global__ __launch_bounds__(256) void k_topk(const float* __restrict__ x,
                                              float* __restrict__ out)
{
  __shared__ float row[Cn];
  __shared__ float red[256];
  long r = blockIdx.x;
  int tid = threadIdx.x;
#pragma unroll
  for (int i = 0; i < 3; ++i) row[tid + i * 256] = x[r * Cn + tid + i * 256];
  __syncthreads();
  float mn = 1e30f, mx = -1e30f;
#pragma unroll
  for (int i = 0; i < 3; ++i) {
    float v = row[tid + i * 256];
    mn = fminf(mn, v); mx = fmaxf(mx, v);
  }
  red[tid] = mn; __syncthreads();
  for (int o = 128; o > 0; o >>= 1) { if (tid < o) red[tid] = fminf(red[tid], red[tid + o]); __syncthreads(); }
  float lo = red[0]; __syncthreads();
  red[tid] = mx; __syncthreads();
  for (int o = 128; o > 0; o >>= 1) { if (tid < o) red[tid] = fmaxf(red[tid], red[tid + o]); __syncthreads(); }
  float hi = red[0] + fmaxf(fabsf(red[0]) * 1e-6f, 1e-6f);
  __syncthreads();
  for (int it = 0; it < 40; ++it) {
    float mid = 0.5f * (lo + hi);
    float c = 0.f;
#pragma unroll
    for (int i = 0; i < 3; ++i) c += (row[tid + i * 256] >= mid) ? 1.f : 0.f;
    red[tid] = c; __syncthreads();
    for (int o = 128; o > 0; o >>= 1) { if (tid < o) red[tid] += red[tid + o]; __syncthreads(); }
    float tot = red[0]; __syncthreads();
    if (tot >= 384.f) lo = mid; else hi = mid;
  }
  float thr = lo;
  float s = 0.f;
#pragma unroll
  for (int i = 0; i < 3; ++i) {
    float v = row[tid + i * 256];
    s += (v >= thr) ? v : 0.f;
  }
  red[tid] = s; __syncthreads();
  for (int o = 128; o > 0; o >>= 1) { if (tid < o) red[tid] += red[tid + o]; __syncthreads(); }
  float S = red[0]; __syncthreads();
  float inv = 1.f / (S + 1e-6f);
  float q = 0.f, sy = 0.f;
#pragma unroll
  for (int i = 0; i < 3; ++i) {
    float v = row[tid + i * 256];
    float y = ((v >= thr) ? v : 0.f) * inv;
    sy += y; q += y * y;
  }
  red[tid] = sy; __syncthreads();
  for (int o = 128; o > 0; o >>= 1) { if (tid < o) red[tid] += red[tid + o]; __syncthreads(); }
  float mean = red[0] / (float)Cn; __syncthreads();
  red[tid] = q; __syncthreads();
  for (int o = 128; o > 0; o >>= 1) { if (tid < o) red[tid] += red[tid + o]; __syncthreads(); }
  float var = (red[0] - (float)Cn * mean * mean) / (float)(Cn - 1);
  float sd = sqrtf(fmaxf(var, 0.f));
#pragma unroll
  for (int i = 0; i < 3; ++i) {
    int c = tid + i * 256;
    float v = row[c];
    float y = ((v >= thr) ? v : 0.f) * inv;
    out[r * Cn + c] = (y - mean) / (sd + 1e-6f);
  }
}

// ---------------------------------------------------------------------------
static void gemm(hipStream_t st,
                 const float* A, long aB, int lda,
                 const float* W, long wB, int ldw, int wT,
                 const float* bias, long biasB,
                 const float* R, long rB, int ldr,
                 float* D, long dB, int ldd,
                 int M, int N, int K, int nb, int ep, const float* s0)
{
  dim3 g((N + 63) / 64, (M + 63) / 64, nb);
  k_gemm<<<g, 128, 0, st>>>(A, aB, lda, W, wB, ldw, wT, bias, biasB,
                            R, rB, ldr, D, dB, ldd, M, N, K, ep, s0);
}

extern "C" void kernel_launch(void* const* d_in, const int* in_sizes, int n_in,
                              void* d_out, int out_size, void* d_ws, size_t ws_size,
                              hipStream_t stream)
{
  (void)in_sizes; (void)n_in; (void)out_size; (void)ws_size;
  const float* x     = (const float*)d_in[0];
  const float* fbk   = (const float*)d_in[1];
  const float* prev  = (const float*)d_in[2];
  const float* ln1g  = (const float*)d_in[3];
  const float* ln1b  = (const float*)d_in[4];
  const float* ln2g  = (const float*)d_in[5];
  const float* ln2b  = (const float*)d_in[6];
  const float* Wattn = (const float*)d_in[7];
  const float* battn = (const float*)d_in[8];
  const float* Wcp   = (const float*)d_in[9];
  const float* bcp   = (const float*)d_in[10];
  const float* relb  = (const float*)d_in[11];
  const float* alpha = (const float*)d_in[12];
  const float* Wlat  = (const float*)d_in[13];
  const float* blat  = (const float*)d_in[14];
  const float* Wfuse = (const float*)d_in[15];
  const float* bfuse = (const float*)d_in[16];
  const float* Wbr   = (const float*)d_in[17];
  const float* bbr   = (const float*)d_in[18];
  const float* Wint  = (const float*)d_in[19];
  const float* bint  = (const float*)d_in[20];
  const float* Wp1   = (const float*)d_in[21];
  const float* bp1   = (const float*)d_in[22];
  const float* pr1   = (const float*)d_in[23];
  const float* Wp2   = (const float*)d_in[24];
  const float* bp2   = (const float*)d_in[25];
  const float* pr2   = (const float*)d_in[26];
  const float* Wih   = (const float*)d_in[27];
  const float* bih   = (const float*)d_in[28];
  const float* Whh   = (const float*)d_in[29];
  const float* bhh   = (const float*)d_in[30];
  const float* Wmod  = (const float*)d_in[31];
  const float* bmod  = (const float*)d_in[32];
  const float* Wf1   = (const float*)d_in[33];
  const float* bf1   = (const float*)d_in[34];
  const float* Wf2   = (const float*)d_in[35];
  const float* bf2   = (const float*)d_in[36];
  const float* Wf3   = (const float*)d_in[37];
  const float* bf3   = (const float*)d_in[38];
  const float* convw = (const float*)d_in[39];
  const float* convb = (const float*)d_in[40];
  const float* stdpv = (const float*)d_in[41];
  const float* metav = (const float*)d_in[42];
  const float* ampv  = (const float*)d_in[43];
  const float* freqv = (const float*)d_in[44];
  const float* phsv  = (const float*)d_in[45];
  const float* muv   = (const float*)d_in[46];
  const float* sgv   = (const float*)d_in[47];
  const float* Wosc  = (const float*)d_in[48];
  const float* bosc  = (const float*)d_in[49];
  float* outp = (float*)d_out;

  float* w = (float*)d_ws;
  const size_t SLOT = (size_t)Bsz * Tn * Cn;
  float* S[11];
  for (int i = 0; i < 11; ++i) S[i] = w + (size_t)i * SLOT;
  float* KERV = w + 11 * SLOT;
  float* KERM = KERV + 2048;
  float* LATB = KERM + (size_t)Tn * Tn;

  // slot assignment (lifetime-based reuse)
  float* attn_in = S[0];
  float* qkv  = S[1];               // 3 slots S1..S3
  float* q    = S[4];
  float* k    = S[5];
  float* v    = S[6];
  float* qmod = S[0];
  float* kmod = S[1];
  float* loc  = S[2];
  float* fsh  = S[3];
  float* fsd  = S[4];
  float* x1   = S[5];
  float* br   = S[6];               // 3 slots S6..S8
  float* x2   = S[0];
  float* xg   = S[1];               // 3 slots S1..S3
  float* gru  = S[4];
  float* fbp  = S[5];
  float* mlpi = S[6];
  float* h4   = S[7];               // 4 slots S7..S10
  float* mlpo = S[1];
  float* mbuf = S[2];
  float* m2b  = S[3];
  float* gat  = S[2];
  float* x3   = S[4];
  float* modb = S[5];
  float* x4   = S[3];

  int EW = (int)((SLOT + 255) / 256);
  long THD = (long)Tn * HDn;

  // 1. attn_in = LN1(x) + feedback
  k_ln<<<Bsz * Tn, 256, 0, stream>>>(x, fbk, ln1g, ln1b, attn_in);
  // 2. qkv
  gemm(stream, attn_in, 0, Cn, Wattn, 0, C3, 0, battn, 0, nullptr, 0, 0,
       qkv, 0, C3, Bsz * Tn, C3, Cn, 1, 0, nullptr);
  // 3. split heads
  k_split<<<EW, 256, 0, stream>>>(qkv, q, k, v);
  // 4. circulant operator from alpha
  k_ker<<<1, 1024, 0, stream>>>(alpha, KERV);
  k_circ<<<(Tn * Tn) / 256, 256, 0, stream>>>(KERV, KERM);
  // 5. latent path
  k_latent<<<Bsz * Hn, 64, 0, stream>>>(v, Wlat, blat, Wfuse, bfuse, LATB);
  // 6. q_mod / k_mod = Ker @ q|k  (batched over B*H)
  gemm(stream, KERM, 0, Tn, q, THD, HDn, 0, nullptr, 0, nullptr, 0, 0,
       qmod, THD, HDn, Tn, HDn, Tn, Bsz * Hn, 0, nullptr);
  gemm(stream, KERM, 0, Tn, k, THD, HDn, 0, nullptr, 0, nullptr, 0, 0,
       kmod, THD, HDn, Tn, HDn, Tn, Bsz * Hn, 0, nullptr);
  // 7. flash attention
  k_attn<<<dim3(Tn / 64, Bsz * Hn), 128, 0, stream>>>(qmod, kmod, v, relb, loc);
  // 8. fused head = local @ Wfuse[:64] + latb (broadcast rows)
  gemm(stream, loc, THD, HDn, Wfuse, 0, HDn, 0, nullptr, 0, LATB, HDn, 0,
       fsh, THD, HDn, Tn, HDn, HDn, Bsz * Hn, 0, nullptr);
  k_perm<<<EW, 256, 0, stream>>>(fsh, fsd);
  // 10. x1 = x + fused @ Wcproj + b
  gemm(stream, fsd, 0, Cn, Wcp, 0, Cn, 0, bcp, 0, x, 0, Cn,
       x1, 0, Cn, Bsz * Tn, Cn, Cn, 1, 0, nullptr);
  // 11. branches (relu), strided into (B,T,3C)
  gemm(stream, x1, 0, Cn, Wbr, (long)Cn * Cn, Cn, 0, bbr, Cn, nullptr, 0, 0,
       br, Cn, C3, Bsz * Tn, Cn, Cn, 3, 1, nullptr);
  // 12. x2 = x1 + br @ Wint + b
  gemm(stream, br, 0, C3, Wint, 0, Cn, 0, bint, 0, x1, 0, Cn,
       x2, 0, Cn, Bsz * Tn, Cn, C3, 1, 0, nullptr);
  // 13. GRU input gates, 14. recurrence, 15. fb_proc = gru * sigmoid(modl)
  gemm(stream, fbk, 0, Cn, Wih, 0, Cn, 1, bih, 0, nullptr, 0, 0,
       xg, 0, C3, Bsz * Tn, C3, Cn, 1, 0, nullptr);
  k_gru<<<Bsz, 1024, 0, stream>>>(xg, Whh, bhh, gru);
  gemm(stream, fbk, 0, Cn, Wmod, 0, Cn, 0, bmod, 0, gru, 0, Cn,
       fbp, 0, Cn, Bsz * Tn, Cn, Cn, 1, 6, nullptr);
  // 16. mlp_in = LN2(x2)
  k_ln<<<Bsz * Tn, 256, 0, stream>>>(x2, nullptr, ln2g, ln2b, mlpi);
  // 17/18. plastic MLP
  gemm(stream, mlpi, 0, Cn, Wp1, 0, C4, 0, bp1, 0, nullptr, 0, 0,
       h4, 0, C4, Bsz * Tn, C4, Cn, 1, 5, pr1);
  gemm(stream, h4, 0, C4, Wp2, 0, Cn, 0, bp2, 0, nullptr, 0, 0,
       mlpo, 0, Cn, Bsz * Tn, Cn, C4, 1, 4, pr2);
  // 19-22. gating stack (W_f1 split over the concat)
  gemm(stream, mlpi, 0, Cn, Wf1, 0, Cn, 0, nullptr, 0, nullptr, 0, 0,
       mbuf, 0, Cn, Bsz * Tn, Cn, Cn, 1, 0, nullptr);
  gemm(stream, fbp, 0, Cn, Wf1 + (size_t)Cn * Cn, 0, Cn, 0, bf1, 0, mbuf, 0, Cn,
       mbuf, 0, Cn, Bsz * Tn, Cn, Cn, 1, 3, nullptr);
  gemm(stream, mbuf, 0, Cn, Wf2, 0, Cn, 0, bf2, 0, nullptr, 0, 0,
       m2b, 0, Cn, Bsz * Tn, Cn, Cn, 1, 3, nullptr);
  gemm(stream, m2b, 0, Cn, Wf3, 0, Cn, 0, bf3, 0, nullptr, 0, 0,
       gat, 0, Cn, Bsz * Tn, Cn, Cn, 1, 2, nullptr);
  // 23. STDP gate + residual
  k_stdp<<<EW, 256, 0, stream>>>(x2, mlpi, prev, mlpo, gat, convw, convb,
                                 stdpv, metav, x3);
  // 24/25. oscillator
  k_osc<<<EW, 256, 0, stream>>>(x3, ampv, freqv, phsv, muv, sgv, modb);
  gemm(stream, modb, 0, Cn, Wosc, 0, Cn, 0, bosc, 0, x3, 0, Cn,
       x4, 0, Cn, Bsz * Tn, Cn, Cn, 1, 0, nullptr);
  // 26. top-k lateral inhibition + standardize
  k_topk<<<Bsz * Tn, 256, 0, stream>>>(x4, outp);
}
